// BeamRNNT_81853486727328
// MI455X (gfx1250) — matrix operand order repaired and checked by reading, not compiled, not run on hardware
//
#include <hip/hip_runtime.h>
#include <hip/hip_bf16.h>

#define BATCH   8
#define BEAM    5
#define TFRAMES 64
#define MAXSYM  3
#define HID     512
#define ENCD    512
#define VOCABSZ 4096
#define NROW    40      // BATCH*BEAM
#define NPAD    48      // padded to 3 WMMA row tiles
#define LHYP    193     // 1 + TFRAMES*MAXSYM
#define NTILE   4       // N-tiles per wave (A-fragment reuse)

typedef __attribute__((ext_vector_type(16))) __bf16  v16bf;
typedef __attribute__((ext_vector_type(8)))  float   v8f;
typedef __attribute__((ext_vector_type(4)))  unsigned u32x4;

union FragU { v16bf v; u32x4 q[2]; };

__device__ __forceinline__ unsigned short f32_to_bf16(float f){
  union { float f; unsigned u; } c; c.f = f;
  unsigned u = c.u;
  u += 0x7fffu + ((u >> 16) & 1u);           // round-to-nearest-even
  return (unsigned short)(u >> 16);
}
__device__ __forceinline__ float sigm(float x){ return 1.0f / (1.0f + expf(-x)); }

// ---------------------------------------------------------------------------
// Fragment loads for one K-chunk: A (2x b128) + NTILE B tiles (2x b128 each).
// ---------------------------------------------------------------------------
__device__ __forceinline__ void load_frags(const unsigned short* __restrict__ arow,
                                           const unsigned short* __restrict__ bp,
                                           int numKT, int kc, int lane, int half,
                                           FragU& fa, FragU fb[NTILE]){
  const u32x4* pa = (const u32x4*)(arow + kc*32 + half*16);
  fa.q[0] = pa[0];          // k .. k+7
  fa.q[1] = pa[1];          // k+8 .. k+15
  #pragma unroll
  for (int t = 0; t < NTILE; ++t){
    const u32x4* pb = (const u32x4*)(bp + ((size_t)t*numKT + kc)*512 + lane*16);
    fb[t].q[0] = pb[0];
    fb[t].q[1] = pb[1];
  }
}

// ---------------------------------------------------------------------------
// WMMA inner loop: 4-way N-tile blocking + software double buffering.
// Steady state issues chunk(kc+1) loads BEFORE chunk(kc) wmmas so the
// v_wmma ops overlap the in-flight global_load_b128s (staggered
// s_wait_loadcnt instead of 0x0 per mma). unroll-2 turns the ping-pong
// copies into pure register rotation.
// ---------------------------------------------------------------------------
__device__ __forceinline__ void wmma_kloop4(const unsigned short* __restrict__ arow,
                                            const unsigned short* __restrict__ bp,
                                            int numKT, int lane, v8f acc[NTILE]){
  const int half = lane >> 4;
  FragU faC, fbC[NTILE], faN, fbN[NTILE];
  load_frags(arow, bp, numKT, 0, lane, half, faC, fbC);
  #pragma unroll 2
  for (int kc = 0; kc < numKT - 1; ++kc){
    load_frags(arow, bp, numKT, kc + 1, lane, half, faN, fbN);
    #pragma unroll
    for (int t = 0; t < NTILE; ++t){
      acc[t] = __builtin_amdgcn_wmma_f32_16x16x32_bf16(false, faC.v, false, fbC[t].v,
                                                       (short)0, acc[t], false, false);
    }
    faC = faN;
    #pragma unroll
    for (int t = 0; t < NTILE; ++t) fbC[t] = fbN[t];
  }
  #pragma unroll
  for (int t = 0; t < NTILE; ++t){
    acc[t] = __builtin_amdgcn_wmma_f32_16x16x32_bf16(false, faC.v, false, fbC[t].v,
                                                     (short)0, acc[t], false, false);
  }
}

// ---------------------------------------------------------------------------
// Preprocessing: pack f32 weight (K x N row-major) into bf16 WMMA B-fragments.
// Packed index: (((nt*numKT + kc)*32 + lane)*16 + e)
//   value = W[(kc*32 + (lane>>4)*16 + e) * N + nt*16 + (lane&15)]
// ---------------------------------------------------------------------------
__global__ void k_pack_w(const float* __restrict__ W, unsigned short* __restrict__ out,
                         int K, int N){
  int idx = blockIdx.x*blockDim.x + threadIdx.x;
  if (idx >= K*N) return;
  int e    = idx & 15;
  int lane = (idx >> 4) & 31;
  int tile = idx >> 9;
  int numKT = K >> 5;
  int nt = tile / numKT;
  int kc = tile - nt*numKT;
  int k  = kc*32 + (lane >> 4)*16 + e;
  int n  = nt*16 + (lane & 15);
  out[idx] = f32_to_bf16(W[(size_t)k*N + n]);
}

__global__ void k_f32_to_bf16(const float* __restrict__ in, unsigned short* __restrict__ out, int n){
  int i = blockIdx.x*blockDim.x + threadIdx.x;
  if (i < n) out[i] = f32_to_bf16(in[i]);
}

__global__ void k_init_scores(float* scores){
  int r = threadIdx.x;
  if (r < NROW) scores[r] = ((r % BEAM) == 0) ? 0.0f : -INFINITY;
}

// ---------------------------------------------------------------------------
// Generic GEMM: C[M x N] = A(bf16, ld=K) x Bpacked   (one wave, 4 N-tiles)
// ---------------------------------------------------------------------------
__global__ void k_gemm_plain(const unsigned short* __restrict__ A,
                             const unsigned short* __restrict__ Bp,
                             float* __restrict__ C, int K, int N){
  int lane = threadIdx.x, ntb = blockIdx.x, mt = blockIdx.y;
  int numKT = K >> 5;
  v8f acc[NTILE] = {};
  const unsigned short* arow = A + (size_t)(mt*16 + (lane & 15))*K;
  wmma_kloop4(arow, Bp + (size_t)(ntb*NTILE)*numKT*512, numKT, lane, acc);
  int m0 = mt*16 + (lane >> 4)*8;
  #pragma unroll
  for (int t = 0; t < NTILE; ++t){
    int n = (ntb*NTILE + t)*16 + (lane & 15);
    #pragma unroll
    for (int e = 0; e < 8; ++e) C[(size_t)(m0+e)*N + n] = acc[t][e];
  }
}

// gates: g = emb[tok] @ WxP + h @ WhP + b         (NPAD x 2048)
__global__ void k_gemm_gates(const unsigned short* __restrict__ emb_bf,
                             const int* __restrict__ tok,
                             const unsigned short* __restrict__ h_bf,
                             const unsigned short* __restrict__ WxP,
                             const unsigned short* __restrict__ WhP,
                             const float* __restrict__ bias,
                             float* __restrict__ g){
  int lane = threadIdx.x, ntb = blockIdx.x, mt = blockIdx.y;
  const int numKT = HID >> 5;
  int m = mt*16 + (lane & 15);
  v8f acc[NTILE] = {};
  const unsigned short* xrow = emb_bf + (size_t)tok[m]*HID;
  wmma_kloop4(xrow, WxP + (size_t)(ntb*NTILE)*numKT*512, numKT, lane, acc);
  const unsigned short* hrow = h_bf + (size_t)m*HID;
  wmma_kloop4(hrow, WhP + (size_t)(ntb*NTILE)*numKT*512, numKT, lane, acc);
  int m0 = mt*16 + (lane >> 4)*8;
  #pragma unroll
  for (int t = 0; t < NTILE; ++t){
    int n = (ntb*NTILE + t)*16 + (lane & 15);
    float bn = bias[n];
    #pragma unroll
    for (int e = 0; e < 8; ++e) g[(size_t)(m0+e)*(4*HID) + n] = acc[t][e] + bn;
  }
}

// LSTM elementwise: h2,c2 from gates (rows 0..39)
__global__ void k_lstm(const float* __restrict__ g, const float* __restrict__ cCur,
                       float* __restrict__ h2f, float* __restrict__ c2f,
                       unsigned short* __restrict__ h2bf){
  int r = blockIdx.x, tid = threadIdx.x;
  #pragma unroll
  for (int k = 0; k < 2; ++k){
    int c = tid + k*256;
    const float* gr = g + (size_t)r*(4*HID);
    float i_ = gr[c], f_ = gr[HID + c], gg = gr[2*HID + c], o_ = gr[3*HID + c];
    float cold = cCur[(size_t)r*HID + c];
    float c2 = sigm(f_)*cold + sigm(i_)*tanhf(gg);
    float h2 = sigm(o_)*tanhf(c2);
    h2f[(size_t)r*HID + c] = h2;
    c2f[(size_t)r*HID + c] = c2;
    h2bf[(size_t)r*HID + c] = f32_to_bf16(h2);
  }
}

// joint: jt = tanh(encProj(frame) + h2 @ WdP + bj)   -> bf16 (NPAD x 512)
__global__ void k_gemm_joint(const unsigned short* __restrict__ h2bf,
                             const unsigned short* __restrict__ WdP,
                             const float* __restrict__ P,   // (B*T, HID) enc@We
                             const float* __restrict__ bj,
                             int frame, unsigned short* __restrict__ jt){
  int lane = threadIdx.x, ntb = blockIdx.x, mt = blockIdx.y;
  const int numKT = HID >> 5;
  v8f acc[NTILE] = {};
  const unsigned short* arow = h2bf + (size_t)(mt*16 + (lane & 15))*HID;
  wmma_kloop4(arow, WdP + (size_t)(ntb*NTILE)*numKT*512, numKT, lane, acc);
  int m0 = mt*16 + (lane >> 4)*8;
  #pragma unroll
  for (int t = 0; t < NTILE; ++t){
    int n = (ntb*NTILE + t)*16 + (lane & 15);
    float bjn = bj[n];
    #pragma unroll
    for (int e = 0; e < 8; ++e){
      int m  = m0 + e;
      int bb = (m < NROW) ? (m / BEAM) : (BATCH - 1);
      float val = tanhf(acc[t][e] + P[((size_t)bb*TFRAMES + frame)*HID + n] + bjn);
      jt[(size_t)m*HID + n] = f32_to_bf16(val);
    }
  }
}

// logits: jt @ WoP + bo   (NPAD x 4096)
__global__ void k_gemm_logits(const unsigned short* __restrict__ jt,
                              const unsigned short* __restrict__ WoP,
                              const float* __restrict__ bo,
                              float* __restrict__ logits){
  int lane = threadIdx.x, ntb = blockIdx.x, mt = blockIdx.y;
  const int numKT = HID >> 5;
  v8f acc[NTILE] = {};
  const unsigned short* arow = jt + (size_t)(mt*16 + (lane & 15))*HID;
  wmma_kloop4(arow, WoP + (size_t)(ntb*NTILE)*numKT*512, numKT, lane, acc);
  int m0 = mt*16 + (lane >> 4)*8;
  #pragma unroll
  for (int t = 0; t < NTILE; ++t){
    int n = (ntb*NTILE + t)*16 + (lane & 15);
    float bon = bo[n];
    #pragma unroll
    for (int e = 0; e < 8; ++e) logits[(size_t)(m0+e)*VOCABSZ + n] = acc[t][e] + bon;
  }
}

// log_softmax + top-5 per row (40 rows of 4096)
__global__ void k_softmax_top5(const float* __restrict__ logits,
                               float* __restrict__ vals, int* __restrict__ idxs){
  int r = blockIdx.x, tid = threadIdx.x;
  __shared__ float sv[256];
  __shared__ int   si[256];
  __shared__ int   winner;
  const float* row = logits + (size_t)r*VOCABSZ;
  float v[16];
  #pragma unroll
  for (int k = 0; k < 16; ++k) v[k] = row[tid + (k << 8)];
  float m = -INFINITY;
  #pragma unroll
  for (int k = 0; k < 16; ++k) m = fmaxf(m, v[k]);
  sv[tid] = m; __syncthreads();
  for (int s = 128; s > 0; s >>= 1){ if (tid < s) sv[tid] = fmaxf(sv[tid], sv[tid+s]); __syncthreads(); }
  float rowmax = sv[0]; __syncthreads();
  float sum = 0.0f;
  #pragma unroll
  for (int k = 0; k < 16; ++k) sum += expf(v[k] - rowmax);
  sv[tid] = sum; __syncthreads();
  for (int s = 128; s > 0; s >>= 1){ if (tid < s) sv[tid] += sv[tid+s]; __syncthreads(); }
  float logZ = rowmax + logf(sv[0]); __syncthreads();

  unsigned mask = 0;
  for (int t = 0; t < BEAM; ++t){
    float bm = -INFINITY; int bi = 0x7fffffff;
    #pragma unroll
    for (int k = 0; k < 16; ++k){
      if (!(mask & (1u << k))){
        int c = tid + (k << 8);
        float x = v[k];
        if (x > bm || (x == bm && c < bi)){ bm = x; bi = c; }
      }
    }
    sv[tid] = bm; si[tid] = bi; __syncthreads();
    for (int s = 128; s > 0; s >>= 1){
      if (tid < s){
        if (sv[tid+s] > sv[tid] || (sv[tid+s] == sv[tid] && si[tid+s] < si[tid])){
          sv[tid] = sv[tid+s]; si[tid] = si[tid+s];
        }
      }
      __syncthreads();
    }
    if (tid == 0){
      winner = si[0];
      vals[r*BEAM + t] = sv[0] - logZ;
      idxs[r*BEAM + t] = si[0];
    }
    __syncthreads();
    int w = winner;
    if ((w & 255) == tid) mask |= 1u << (w >> 8);
    __syncthreads();
  }
}

// beam combine per batch: top-5 of 25 with end/length masking
__global__ void k_combine(const float* __restrict__ vals, const int* __restrict__ idxs,
                          float* __restrict__ scores, int* __restrict__ end_flag,
                          const int* __restrict__ enc_len, int frame, int sym,
                          int* __restrict__ sel_src, int* __restrict__ sel_tok,
                          int* __restrict__ hyp_col){
  int b = blockIdx.x;
  if (threadIdx.x != 0) return;
  float cand[BEAM*BEAM]; int ctok[BEAM*BEAM]; bool used[BEAM*BEAM];
  bool efold[BEAM];
  int elen = enc_len[b];
  for (int s = 0; s < BEAM; ++s){
    int r = b*BEAM + s;
    bool ef = (sym == 0) ? false : (end_flag[r] != 0);
    efold[s] = ef;
    bool flag = ef || (elen <= frame);
    float sc = scores[r];
    for (int c = 0; c < BEAM; ++c){
      float vv = vals[r*BEAM + c];
      int   id = idxs[r*BEAM + c];
      if (flag){ vv = (c == 0) ? 0.0f : -INFINITY; id = 0; }
      cand[s*BEAM + c] = sc + vv;
      ctok[s*BEAM + c] = id;
      used[s*BEAM + c] = false;
    }
  }
  for (int t = 0; t < BEAM; ++t){
    int gi = -1; float bv = -INFINITY;
    for (int q = 0; q < BEAM*BEAM; ++q){
      if (!used[q] && (gi < 0 || cand[q] > bv)){ bv = cand[q]; gi = q; }
    }
    used[gi] = true;
    int r    = b*BEAM + t;
    int src  = gi / BEAM;
    int tokn = ctok[gi];
    scores[r] = bv;
    bool efn = efold[t] || (tokn == 0);       // reference: end_flag |= (tok==BLANK), un-gathered
    end_flag[r] = efn ? 1 : 0;
    sel_src[r]  = b*BEAM + src;
    sel_tok[r]  = tokn;
    hyp_col[r]  = (efn || (elen <= frame)) ? 0 : tokn;
  }
}

// gather states + hypotheses into ping-pong buffers, derive cur_tok
__global__ void k_gather(const float* __restrict__ hOld, const float* __restrict__ cOld,
                         const float* __restrict__ h2f,  const float* __restrict__ c2f,
                         float* __restrict__ hNew, float* __restrict__ cNew,
                         unsigned short* __restrict__ hbfNew,
                         const int* __restrict__ hypsOld, int* __restrict__ hypsNew,
                         const int* __restrict__ sel_src, const int* __restrict__ sel_tok,
                         const int* __restrict__ hyp_col, const int* __restrict__ end_flag,
                         int* __restrict__ tok, int col){
  int r = blockIdx.x, tid = threadIdx.x;
  int src = sel_src[r];
  bool ef = end_flag[r] != 0;
  const float* hs = ef ? (hOld + (size_t)src*HID) : (h2f + (size_t)src*HID);
  const float* cs = ef ? (cOld + (size_t)src*HID) : (c2f + (size_t)src*HID);
  for (int c = tid; c < HID; c += 64){
    float hv = hs[c];
    hNew[(size_t)r*HID + c]  = hv;
    hbfNew[(size_t)r*HID + c] = f32_to_bf16(hv);
    cNew[(size_t)r*HID + c]  = cs[c];
  }
  for (int p = tid; p < LHYP; p += 64){
    int v = hypsOld[(size_t)src*LHYP + p];
    hypsNew[(size_t)r*LHYP + p] = (p == col) ? hyp_col[r] : v;
  }
  if (tid == 0){
    int prev = 0;
    for (int p = LHYP - 1; p >= 0; --p){
      int v = hypsOld[(size_t)src*LHYP + p];
      if (v != 0){ prev = v; break; }
    }
    tok[r] = ef ? prev : sel_tok[r];
  }
}

__global__ void k_finalize(const float* __restrict__ scores, const int* __restrict__ hyps,
                           float* __restrict__ out){
  __shared__ int best[BATCH]; __shared__ float bsc[BATCH];
  int tid = threadIdx.x;
  if (tid < BATCH){
    int b = tid; float bv = scores[b*BEAM]; int bi = 0;
    for (int s = 1; s < BEAM; ++s){ float v = scores[b*BEAM + s]; if (v > bv){ bv = v; bi = s; } }
    best[b] = b*BEAM + bi; bsc[b] = bv;
  }
  __syncthreads();
  for (int idx = tid; idx < BATCH*LHYP; idx += 256){
    int b = idx / LHYP, p = idx % LHYP;
    out[idx] = (float)hyps[(size_t)best[b]*LHYP + p];
  }
  if (tid < BATCH) out[BATCH*LHYP + tid] = bsc[tid];
}

// ---------------------------------------------------------------------------
extern "C" void kernel_launch(void* const* d_in, const int* in_sizes, int n_in,
                              void* d_out, int out_size, void* d_ws, size_t ws_size,
                              hipStream_t stream) {
  (void)in_sizes; (void)n_in; (void)out_size; (void)ws_size;
  const float* enc     = (const float*)d_in[0];
  const int*   enc_len = (const int*)  d_in[1];
  const float* emb     = (const float*)d_in[2];
  const float* Wx      = (const float*)d_in[3];
  const float* Wh      = (const float*)d_in[4];
  const float* bvec    = (const float*)d_in[5];
  const float* We      = (const float*)d_in[6];
  const float* Wd      = (const float*)d_in[7];
  const float* bj      = (const float*)d_in[8];
  const float* Wo      = (const float*)d_in[9];
  const float* bo      = (const float*)d_in[10];

  char* ws = (char*)d_ws; size_t off = 0;
  auto carve = [&](size_t bytes)->char*{
    char* p = ws + off;
    off = (off + bytes + 255) & ~(size_t)255;
    return p;
  };
  unsigned short* WxP    = (unsigned short*)carve((size_t)HID*4*HID*2);
  unsigned short* WhP    = (unsigned short*)carve((size_t)HID*4*HID*2);
  unsigned short* WdP    = (unsigned short*)carve((size_t)HID*HID*2);
  unsigned short* WoP    = (unsigned short*)carve((size_t)HID*VOCABSZ*2);
  unsigned short* WeP    = (unsigned short*)carve((size_t)ENCD*HID*2);
  unsigned short* emb_bf = (unsigned short*)carve((size_t)VOCABSZ*HID*2);
  unsigned short* enc_bf = (unsigned short*)carve((size_t)BATCH*TFRAMES*ENCD*2);
  float*          P      = (float*)carve((size_t)BATCH*TFRAMES*HID*4);
  float*          g      = (float*)carve((size_t)NPAD*4*HID*4);
  float*          hA     = (float*)carve((size_t)NPAD*HID*4);
  float*          hBuf   = (float*)carve((size_t)NPAD*HID*4);
  float*          cA     = (float*)carve((size_t)NPAD*HID*4);
  float*          cBuf   = (float*)carve((size_t)NPAD*HID*4);
  unsigned short* hbfA   = (unsigned short*)carve((size_t)NPAD*HID*2);
  unsigned short* hbfB   = (unsigned short*)carve((size_t)NPAD*HID*2);
  float*          h2f    = (float*)carve((size_t)NPAD*HID*4);
  float*          c2f    = (float*)carve((size_t)NPAD*HID*4);
  unsigned short* h2bf   = (unsigned short*)carve((size_t)NPAD*HID*2);
  unsigned short* jt     = (unsigned short*)carve((size_t)NPAD*HID*2);
  float*          logits = (float*)carve((size_t)NPAD*VOCABSZ*4);
  float*          vals   = (float*)carve((size_t)NROW*BEAM*4);
  int*            idxs   = (int*)carve((size_t)NROW*BEAM*4);
  float*          scores = (float*)carve((size_t)NROW*4);
  int*            eflag  = (int*)carve((size_t)NROW*4);
  int*            selsrc = (int*)carve((size_t)NROW*4);
  int*            seltok = (int*)carve((size_t)NROW*4);
  int*            hypcol = (int*)carve((size_t)NROW*4);
  int*            tok    = (int*)carve((size_t)NPAD*4);
  int*            hyA    = (int*)carve((size_t)NROW*LHYP*4);
  int*            hyB    = (int*)carve((size_t)NROW*LHYP*4);

  // ---- one-time (per launch) preprocessing ----
  hipMemsetAsync(hA,   0, (size_t)NPAD*HID*4, stream);
  hipMemsetAsync(hBuf, 0, (size_t)NPAD*HID*4, stream);
  hipMemsetAsync(cA,   0, (size_t)NPAD*HID*4, stream);
  hipMemsetAsync(cBuf, 0, (size_t)NPAD*HID*4, stream);
  hipMemsetAsync(hbfA, 0, (size_t)NPAD*HID*2, stream);
  hipMemsetAsync(hbfB, 0, (size_t)NPAD*HID*2, stream);
  hipMemsetAsync(h2bf, 0, (size_t)NPAD*HID*2, stream);
  hipMemsetAsync(hyA,  0, (size_t)NROW*LHYP*4, stream);
  hipMemsetAsync(hyB,  0, (size_t)NROW*LHYP*4, stream);
  hipMemsetAsync(tok,  0, (size_t)NPAD*4, stream);
  hipMemsetAsync(eflag,0, (size_t)NROW*4, stream);
  k_init_scores<<<1, 64, 0, stream>>>(scores);

  k_pack_w<<<((HID*4*HID)+255)/256, 256, 0, stream>>>(Wx, WxP, HID, 4*HID);
  k_pack_w<<<((HID*4*HID)+255)/256, 256, 0, stream>>>(Wh, WhP, HID, 4*HID);
  k_pack_w<<<((HID*HID)+255)/256,   256, 0, stream>>>(Wd, WdP, HID, HID);
  k_pack_w<<<((HID*VOCABSZ)+255)/256,256,0, stream>>>(Wo, WoP, HID, VOCABSZ);
  k_pack_w<<<((ENCD*HID)+255)/256,  256, 0, stream>>>(We, WeP, ENCD, HID);
  k_f32_to_bf16<<<((VOCABSZ*HID)+255)/256, 256, 0, stream>>>(emb, emb_bf, VOCABSZ*HID);
  k_f32_to_bf16<<<((BATCH*TFRAMES*ENCD)+255)/256, 256, 0, stream>>>(enc, enc_bf, BATCH*TFRAMES*ENCD);

  // encoder projection for all frames: P = enc_bf @ WeP  (M=512, K=512, N=512)
  k_gemm_plain<<<dim3(HID/(16*NTILE), (BATCH*TFRAMES)/16), 32, 0, stream>>>(enc_bf, WeP, P, ENCD, HID);

  // ---- sequential decode: 64 frames x 3 symbols ----
  float* hC = hA;  float* hN = hBuf;
  float* cC = cA;  float* cN = cBuf;
  unsigned short* hbfC = hbfA; unsigned short* hbfN = hbfB;
  int* hyC = hyA;  int* hyN = hyB;

  for (int i = 0; i < TFRAMES; ++i){
    for (int j = 0; j < MAXSYM; ++j){
      k_gemm_gates <<<dim3((4*HID)/(16*NTILE), NPAD/16), 32, 0, stream>>>(emb_bf, tok, hbfC, WxP, WhP, bvec, g);
      k_lstm       <<<NROW, 256, 0, stream>>>(g, cC, h2f, c2f, h2bf);
      k_gemm_joint <<<dim3(HID/(16*NTILE), NPAD/16), 32, 0, stream>>>(h2bf, WdP, P, bj, i, jt);
      k_gemm_logits<<<dim3(VOCABSZ/(16*NTILE), NPAD/16), 32, 0, stream>>>(jt, WoP, bo, logits);
      k_softmax_top5<<<NROW, 256, 0, stream>>>(logits, vals, idxs);
      k_combine    <<<BATCH, 32, 0, stream>>>(vals, idxs, scores, eflag, enc_len, i, j,
                                              selsrc, seltok, hypcol);
      k_gather     <<<NROW, 64, 0, stream>>>(hC, cC, h2f, c2f, hN, cN, hbfN,
                                             hyC, hyN, selsrc, seltok, hypcol, eflag,
                                             tok, 1 + i*MAXSYM + j);
      { float* t = hC; hC = hN; hN = t; }
      { float* t = cC; cC = cN; cN = t; }
      { unsigned short* t = hbfC; hbfC = hbfN; hbfN = t; }
      { int* t = hyC; hyC = hyN; hyN = t; }
    }
  }

  k_finalize<<<1, 256, 0, stream>>>(scores, hyC, (float*)d_out);
}
